// Attention_6107443495515
// MI455X (gfx1250) — matrix operand order repaired and checked by reading, not compiled, not run on hardware
//
#include <hip/hip_runtime.h>
#include <hip/hip_bf16.h>

// CDNA5 / gfx1250, wave32. All matmuls via v_wmma_f32_16x16x32_bf16.

typedef __attribute__((ext_vector_type(16))) __bf16 v16bf;
typedef __attribute__((ext_vector_type(8)))  float  v8f;

union FragBF {
    v16bf v;
    unsigned int u[8];
};

__device__ __forceinline__ unsigned short f2bf(float x) {
    unsigned int u = __float_as_uint(x);
    u += 0x7FFFu + ((u >> 16) & 1u);       // round-to-nearest-even
    return (unsigned short)(u >> 16);
}

#define S_LEN 2048
#define NH    32
#define NKV   8
#define HD    64

// ---------------------------------------------------------------------------
// GEMM: C = A(MxK) * B(KxN).  A is f32 or bf16 (ABF16), B is f32.
// Block: 256 threads (8 waves), tile 128x64, K-step 32.
// Wave grid 4x2; each wave owns a 32x32 C slice = 4 WMMA tiles
// (2 A-frags x 2 B-frags, operands reused across the 4 WMMAs).
// Epilogue MODE: 0 = RoPE -> Q[B,NH,S,HD] bf16
//                1 = RoPE -> K[B,NKV,S,HD] bf16
//                2 = V^T  -> V[B,NKV,HD,S] bf16
//                3 = f32 row-major out (ldc = Nmat)
// MODE/ABF16 are template params so the staged-load loop is straight-line
// (the previous revision branched per element on a runtime flag).
// ---------------------------------------------------------------------------
template<int MODE, bool ABF16>
__global__ __launch_bounds__(256) void gemm_wmma(
    const void* __restrict__ Aptr,
    const float* __restrict__ Bm, int Kdim, int Nmat,
    unsigned short* __restrict__ outb, float* __restrict__ outf)
{
    __shared__ unsigned short As[128][34];  // row-major [m][k], bf16
    __shared__ unsigned short Bs[64][34];   // transposed [n][k], bf16

    const int t    = threadIdx.x;
    const int lane = t & 31;
    const int wave = t >> 5;
    const int wr   = wave >> 1;             // 0..3 -> M sub-tile of 32
    const int wc   = wave & 1;              // 0..1 -> N sub-tile of 32
    const int hh   = lane >> 4;             // lane-half (0/1)
    const int l16  = lane & 15;
    const int m0   = blockIdx.y * 128;
    const int n0   = blockIdx.x * 64;

    const float*          Af = (const float*)Aptr;
    const unsigned short* Ab = (const unsigned short*)Aptr;

    v8f acc[2][2] = {};                     // [a-tile][b-tile]

    const int kba = hh ? 8  : 0;            // A-fragment K base per lane-half
    const int kbb = hh ? 8  : 0;            // B-fragment K base per lane-half

    for (int k0 = 0; k0 < Kdim; k0 += 32) {
        // prefetch next K-tile (global_prefetch_b8) while this one is staged
        if (k0 + 32 < Kdim) {
            if (ABF16)
                __builtin_prefetch(Ab + (size_t)(m0 + (t >> 4)) * Kdim +
                                   k0 + 32 + (t & 15) * 2, 0, 0);
            else
                __builtin_prefetch(Af + (size_t)(m0 + (t >> 4)) * Kdim +
                                   k0 + 32 + (t & 15) * 2, 0, 0);
            __builtin_prefetch(Bm + (size_t)(k0 + 32 + 2 * (t >> 6)) * Nmat +
                               n0 + (t & 63), 0, 0);
        }
        __syncthreads();
        // ---- stage A tile (128x32) into LDS as bf16, packed pairs ----
        #pragma unroll
        for (int e = 0; e < 8; ++e) {
            int idx = e * 256 + t;          // pair index, 2048 pairs
            int r   = idx >> 4;
            int c   = (idx & 15) * 2;
            unsigned int pack;
            if (ABF16) {
                pack = *(const unsigned int*)(Ab + (size_t)(m0 + r) * Kdim + k0 + c);
            } else {
                const float* p = Af + (size_t)(m0 + r) * Kdim + k0 + c;
                pack = (unsigned int)f2bf(p[0]) | ((unsigned int)f2bf(p[1]) << 16);
            }
            *(unsigned int*)&As[r][c] = pack;
        }
        // ---- stage B tile (32x64) transposed into LDS [n][k] ----
        #pragma unroll
        for (int e = 0; e < 4; ++e) {
            int idx = e * 256 + t;
            int kp  = idx >> 6;             // 0..15 (k pair)
            int n   = idx & 63;
            const float* p = Bm + (size_t)(k0 + 2 * kp) * Nmat + n0 + n;
            unsigned int pack = (unsigned int)f2bf(p[0]) |
                                ((unsigned int)f2bf(p[Nmat]) << 16);
            *(unsigned int*)&Bs[n][2 * kp] = pack;
        }
        __syncthreads();

        // ---- fragments (ISA 16-bit WMMA layouts) ----
        FragBF afr[2], bfr[2];
        #pragma unroll
        for (int at = 0; at < 2; ++at) {
            const int arow = wr * 32 + at * 16 + l16;
            #pragma unroll
            for (int v = 0; v < 8; ++v) {
                int kk = kba + (v < 4 ? v * 2 : 16 + (v - 4) * 2);
                afr[at].u[v] = *(const unsigned int*)&As[arow][kk];
            }
        }
        #pragma unroll
        for (int nt = 0; nt < 2; ++nt) {
            const int brow = wc * 32 + nt * 16 + l16;
            #pragma unroll
            for (int v = 0; v < 8; ++v) {
                int kk = kbb + (v < 4 ? v * 2 : 16 + (v - 4) * 2);
                bfr[nt].u[v] = *(const unsigned int*)&Bs[brow][kk];
            }
        }
        #pragma unroll
        for (int at = 0; at < 2; ++at)
            #pragma unroll
            for (int nt = 0; nt < 2; ++nt)
                acc[at][nt] = __builtin_amdgcn_wmma_f32_16x16x32_bf16(
                    false, afr[at].v, false, bfr[nt].v,
                    (short)0, acc[at][nt], false, false);
    }

    // ---- epilogue ----
    #pragma unroll
    for (int at = 0; at < 2; ++at) {
        #pragma unroll
        for (int nt = 0; nt < 2; ++nt) {
            #pragma unroll
            for (int i = 0; i < 8; ++i) {
                float val = acc[at][nt][i];
                int rowg = m0 + wr * 32 + at * 16 + i + hh * 8;
                int col  = n0 + wc * 32 + nt * 16 + l16;
                if (MODE == 3) {
                    outf[(size_t)rowg * Nmat + col] = val;
                } else {
                    int bi   = rowg >> 11;            // / S_LEN
                    int s    = rowg & (S_LEN - 1);
                    int head = col >> 6;
                    int d    = col & 63;
                    if (MODE == 2) {                  // V^T : [B,NKV,HD,S]
                        outb[(((size_t)(bi * NKV + head)) * HD + d) * S_LEN + s] =
                            f2bf(val);
                    } else {                          // RoPE on Q / K
                        // inv_freq = theta^-((d&~1)/64); ln(10000)=9.2103403...
                        float ang = (float)s *
                            __expf((float)(d & ~1) * (-9.210340371976184f / 64.0f));
                        float cs = __cosf(ang), sn = __sinf(ang);
                        float other = __shfl_xor(val, 1, 32); // even<->odd pair
                        float r = (col & 1) ? (other * sn + val * cs)
                                            : (val * cs - other * sn);
                        size_t idx = (MODE == 0)
                            ? (((size_t)(bi * NH  + head)) * S_LEN + s) * HD + d
                            : (((size_t)(bi * NKV + head)) * S_LEN + s) * HD + d;
                        outb[idx] = f2bf(r);
                    }
                }
            }
        }
    }
}

// ---------------------------------------------------------------------------
// Flash attention: one wave per (b, h, 16-query tile).
// Q:[B,NH,S,HD] bf16, K:[B,NKV,S,HD] bf16, V^T:[B,NKV,HD,S] bf16,
// alibi:[1,NH,S,S] f32, O:[B*S, NH*HD] bf16.
// ---------------------------------------------------------------------------
__global__ __launch_bounds__(32) void attn_wmma(
    const unsigned short* __restrict__ Q,
    const unsigned short* __restrict__ Kb,
    const unsigned short* __restrict__ Vt,
    const float* __restrict__ alibi,
    unsigned short* __restrict__ O)
{
    __shared__ unsigned short Pls[16][34];  // P staging (C-layout -> A-layout)

    const int bid = blockIdx.x;
    const int qt  = bid & 127;              // S/16 = 128 query tiles
    const int h   = (bid >> 7) & 31;
    const int b   = bid >> 12;
    const int q0  = qt * 16;
    const int hk  = h >> 2;                 // GQA: h / N_REP
    const int lane = threadIdx.x;
    const int hh   = lane >> 4;
    const int l16  = lane & 15;
    const int kpa  = hh ? 8  : 0;
    const int kbb  = hh ? 8  : 0;

    // ---- Q A-fragments (2 d-steps of 32) ----
    FragBF qf[2];
    const unsigned short* Qrow =
        Q + (((size_t)(b * NH + h)) * S_LEN + q0 + l16) * HD;
    #pragma unroll
    for (int f = 0; f < 2; ++f) {
        #pragma unroll
        for (int v = 0; v < 8; ++v) {
            int d = f * 32 + kpa + (v < 4 ? v * 2 : 16 + (v - 4) * 2);
            qf[f].u[v] = *(const unsigned int*)(Qrow + d);
        }
    }

    float m[8], l[8];
    #pragma unroll
    for (int i = 0; i < 8; ++i) { m[i] = -1e30f; l[i] = 0.f; }
    v8f o[4] = {};                          // O tile 16q x 64d (4 C frags)

    const float  scale = 0.125f;            // 1/sqrt(64)
    const int    kend  = q0 + 16;           // causal bound (exclusive)
    const size_t kbase = ((size_t)(b * NKV + hk)) * S_LEN;
    const size_t vbase = ((size_t)(b * NKV + hk)) * HD;

    for (int k0 = 0; k0 < kend; k0 += 32) {
        v8f sc0 = {0.f,0.f,0.f,0.f,0.f,0.f,0.f,0.f};
        v8f sc1 = {0.f,0.f,0.f,0.f,0.f,0.f,0.f,0.f};
        // ---- QK^T: two 16-token B tiles x two 32-d steps ----
        #pragma unroll
        for (int nt = 0; nt < 2; ++nt) {
            int kt  = k0 + nt * 16 + l16;
            int ktc = kt < S_LEN ? kt : (S_LEN - 1);
            const unsigned short* Krow = Kb + (kbase + ktc) * HD;
            #pragma unroll
            for (int ds = 0; ds < 2; ++ds) {
                FragBF kf;
                #pragma unroll
                for (int v = 0; v < 8; ++v) {
                    int d = ds * 32 + kbb + (v < 4 ? 2 * v : 16 + 2 * (v - 4));
                    kf.u[v] = *(const unsigned int*)(Krow + d);
                }
                if (nt == 0)
                    sc0 = __builtin_amdgcn_wmma_f32_16x16x32_bf16(false, qf[ds].v,
                          false, kf.v, (short)0, sc0, false, false);
                else
                    sc1 = __builtin_amdgcn_wmma_f32_16x16x32_bf16(false, qf[ds].v,
                          false, kf.v, (short)0, sc1, false, false);
            }
        }
        // ---- scale + alibi + causal mask + online softmax ----
        const int kt0 = k0 + l16;
        const int kt1 = k0 + 16 + l16;
        #pragma unroll
        for (int i = 0; i < 8; ++i) {
            int q = q0 + i + hh * 8;
            const size_t arow = ((size_t)h * S_LEN + q) * S_LEN;
            float s0 = -1e30f, s1 = -1e30f;
            if (kt0 <= q) s0 = sc0[i] * scale + alibi[arow + kt0];
            if (kt1 <= q) s1 = sc1[i] * scale + alibi[arow + kt1];
            float mx = fmaxf(s0, s1);
            #pragma unroll
            for (int off = 1; off < 16; off <<= 1)
                mx = fmaxf(mx, __shfl_xor(mx, off, 32));
            float nm = fmaxf(m[i], mx);
            float p0 = __expf(s0 - nm);
            float p1 = __expf(s1 - nm);
            float rs = p0 + p1;
            #pragma unroll
            for (int off = 1; off < 16; off <<= 1)
                rs += __shfl_xor(rs, off, 32);
            float alpha = __expf(m[i] - nm);
            l[i] = l[i] * alpha + rs;
            m[i] = nm;
            #pragma unroll
            for (int dt = 0; dt < 4; ++dt) o[dt][i] *= alpha;
            Pls[i + hh * 8][l16]      = f2bf(p0);
            Pls[i + hh * 8][16 + l16] = f2bf(p1);
        }
        __syncthreads();
        // ---- P as A-fragment (transposed through LDS) ----
        FragBF pf;
        #pragma unroll
        for (int v = 0; v < 8; ++v) {
            int kk = kpa + (v < 4 ? 2 * v : 16 + 2 * (v - 4));
            pf.u[v] = *(const unsigned int*)&Pls[l16][kk];
        }
        // ---- PV: 4 d-tiles of 16, K-dim = 32 tokens ----
        #pragma unroll
        for (int dt = 0; dt < 4; ++dt) {
            const unsigned short* Vrow = Vt + (vbase + dt * 16 + l16) * S_LEN;
            FragBF vf;
            #pragma unroll
            for (int v = 0; v < 8; ++v) {
                int tk  = k0 + kbb + (v < 4 ? 2 * v : 16 + 2 * (v - 4));
                int tkc = tk <= (S_LEN - 2) ? tk : (S_LEN - 2);
                vf.u[v] = *(const unsigned int*)(Vrow + tkc);
            }
            o[dt] = __builtin_amdgcn_wmma_f32_16x16x32_bf16(false, pf.v, false,
                    vf.v, (short)0, o[dt], false, false);
        }
        __syncthreads();
    }

    // ---- normalize + store O bf16 [B*S, NH*HD] ----
    #pragma unroll
    for (int dt = 0; dt < 4; ++dt) {
        #pragma unroll
        for (int i = 0; i < 8; ++i) {
            float val = o[dt][i] / l[i];
            size_t row = (size_t)b * S_LEN + q0 + i + hh * 8;
            int    col = h * HD + dt * 16 + l16;
            O[row * (NH * HD) + col] = f2bf(val);
        }
    }
}

// ---------------------------------------------------------------------------
extern "C" void kernel_launch(void* const* d_in, const int* in_sizes, int n_in,
                              void* d_out, int out_size, void* d_ws, size_t ws_size,
                              hipStream_t stream) {
    const float* x     = (const float*)d_in[0];   // (2, 2048, 2048)
    const float* alibi = (const float*)d_in[1];   // (1, 32, 2048, 2048)
    const float* Wq    = (const float*)d_in[2];   // (2048, 2048)
    const float* Wk    = (const float*)d_in[3];   // (2048, 512)
    const float* Wv    = (const float*)d_in[4];   // (2048, 512)
    const float* Wo    = (const float*)d_in[5];   // (2048, 2048)
    float* out = (float*)d_out;                   // (2, 2048, 2048)

    // workspace: Q 16MB | K 4MB | V^T 4MB | O 16MB  (40 MB total)
    char* ws = (char*)d_ws;
    unsigned short* Qb  = (unsigned short*)(ws);
    unsigned short* Kbf = (unsigned short*)(ws + ((size_t)16 << 20));
    unsigned short* Vtb = (unsigned short*)(ws + ((size_t)20 << 20));
    unsigned short* Ob  = (unsigned short*)(ws + ((size_t)24 << 20));

    dim3 blk(256);
    // QKV projections (M = B*S = 4096, K = 2048), block tile 128x64
    gemm_wmma<0, false><<<dim3(2048 / 64, 4096 / 128), blk, 0, stream>>>(
        x, Wq, 2048, 2048, Qb, nullptr);
    gemm_wmma<1, false><<<dim3(512 / 64, 4096 / 128), blk, 0, stream>>>(
        x, Wk, 2048, 512, Kbf, nullptr);
    gemm_wmma<2, false><<<dim3(512 / 64, 4096 / 128), blk, 0, stream>>>(
        x, Wv, 2048, 512, Vtb, nullptr);

    // flash attention: B * NH * (S/16) blocks, 1 wave each
    attn_wmma<<<2 * NH * (S_LEN / 16), 32, 0, stream>>>(Qb, Kbf, Vtb, alibi, Ob);

    // output projection: (4096 x 2048) bf16 @ (2048 x 2048) -> f32
    gemm_wmma<3, true><<<dim3(2048 / 64, 4096 / 128), blk, 0, stream>>>(
        Ob, Wo, 2048, 2048, nullptr, out);
}
